// Attention_78993038508349
// MI455X (gfx1250) — matrix operand order repaired and checked by reading, not compiled, not run on hardware
//
#include <hip/hip_runtime.h>
#include <hip/hip_bf16.h>

typedef _Float16 f16;
typedef __attribute__((ext_vector_type(16))) _Float16 v16h;
typedef __attribute__((ext_vector_type(8)))  _Float16 v8h;
typedef __attribute__((ext_vector_type(4)))  _Float16 v4h;
typedef __attribute__((ext_vector_type(4)))  float    v4f;
typedef __attribute__((ext_vector_type(8)))  float    v8f;

#define B_ 2
#define N_ 2048
#define C_ 384
#define H_ 6
#define HD_ 64
#define SCALE_ 0.125f
#define SROW 2052  // 2048 + 4 pad floats -> successive rows step 4 LDS banks

__device__ __forceinline__ v8f wmma16x16x32(v16h a, v16h b, v8f c) {
  // D = A(16x32 f16) * B(32x16 f16) + C(16x16 f32)
  return __builtin_amdgcn_wmma_f32_16x16x32_f16(false, a, false, b, (short)0, c,
                                                false, false);
}

// ---------------------------------------------------------------------------
// Kernel 0: one-shot f32 -> f16 conversion (x, Wqkv, Wproj), vec4.
// ---------------------------------------------------------------------------
__global__ __launch_bounds__(256) void cvt_kernel(const float* __restrict__ src,
                                                  f16* __restrict__ dst,
                                                  int n4) {
  const int i = blockIdx.x * 256 + threadIdx.x;
  if (i < n4) {
    v4f v = *(const v4f*)(src + 4 * (size_t)i);
    v4h h;
    h[0] = (f16)v[0]; h[1] = (f16)v[1]; h[2] = (f16)v[2]; h[3] = (f16)v[3];
    *(v4h*)(dst + 4 * (size_t)i) = h;
  }
}

// ---------------------------------------------------------------------------
// Kernel 1: QKV projection (all-f16 operands, f32 accumulate).
// 4 waves / block, each wave owns one 16-wide output-column tile.
// Epilogue scatters to per-head f16 buffers:
//   q: [B,H,N,HD] (pre-scaled by 1/sqrt(HD))   k: [B,H,N,HD]   v: [B,H,HD,N]
// ---------------------------------------------------------------------------
__global__ __launch_bounds__(128) void qkv_kernel(
    const f16* __restrict__ xh, const f16* __restrict__ wqkvh,
    const float* __restrict__ bqkv, f16* __restrict__ qh, f16* __restrict__ kh,
    f16* __restrict__ vt) {
  const int lane = threadIdx.x & 31;
  const int wave = threadIdx.x >> 5;
  const int row  = lane & 15;
  const int half = lane >> 4;
  const int mt = blockIdx.x;                    // M/16 = 256
  const int nt = blockIdx.y * 4 + wave;         // 1152/16 = 72 col tiles
  const int mrow = mt * 16 + row;
  const int col  = nt * 16 + row;

  const f16* xrow = xh    + (size_t)mrow * C_;
  const f16* wrow = wqkvh + (size_t)col  * C_;

  v8f c = {};
  for (int kb = 0; kb < C_; kb += 32) {
    // A 16x32 f16: lane(row,half), elem e -> k = kb + half*16 + e
    v8h alo = *(const v8h*)(xrow + kb + half * 16);
    v8h ahi = *(const v8h*)(xrow + kb + 8 + half * 16);
    v16h a;
#pragma unroll
    for (int e = 0; e < 8; ++e) { a[e] = alo[e]; a[e + 8] = ahi[e]; }
    // B 32x16 f16: lane(col,half), elem e -> k = kb + half*16 + e
    v16h b = *(const v16h*)(wrow + kb + half * 16);
    c = wmma16x16x32(a, b, c);
  }

  const int   t    = col / 384;       // 0=q 1=k 2=v
  const int   rem  = col - t * 384;
  const int   h    = rem >> 6;
  const int   d    = rem & 63;
  const float bias = bqkv[col];
#pragma unroll
  for (int r = 0; r < 8; ++r) {
    const int m    = mt * 16 + half * 8 + r;   // C/D: vgpr r -> row half*8+r
    const int bb   = m >> 11;
    const int nrow = m & (N_ - 1);
    const float val = c[r] + bias;
    const size_t head = (size_t)(bb * H_ + h);
    if (t == 0)      qh[(head * N_ + nrow) * HD_ + d] = (f16)(val * SCALE_);
    else if (t == 1) kh[(head * N_ + nrow) * HD_ + d] = (f16)val;
    else             vt[(head * HD_ + d) * N_ + nrow] = (f16)val;  // transposed
  }
}

// ---------------------------------------------------------------------------
// Kernel 2: fused attention. One block (4 waves) per (b,h,16-query-row) tile.
// Raw scores for the whole 16x2048 stripe live in LDS (128.25KB of the 320KB
// CDNA5 WGP LDS -> 2 blocks = 8 waves per WGP).
//   phase 1: waves split the 128 key tiles (2 WMMA each) + rpb -> LDS
//   phase 2: 8-way row-segmented max/sum, partials combined through LDS
//   phase 3: normalized attn streamed to d_out with non-temporal vec4 stores
//   phase 4: P.V -- each wave owns a 16-wide d-slice (1 WMMA per 32-key tile)
// ---------------------------------------------------------------------------
__global__ __launch_bounds__(128) void attn_kernel(
    const f16* __restrict__ qh, const f16* __restrict__ kh,
    const f16* __restrict__ vt, const float* __restrict__ rpb,
    float* __restrict__ attn_out, f16* __restrict__ oh) {
  extern __shared__ float smem[];
  float* S       = smem;                    // [16][SROW]
  float* pmax    = smem + 16 * SROW;        // [8][16]
  float* psum    = pmax + 8 * 16;           // [8][16]
  float* maxbuf  = psum + 8 * 16;           // [16]
  float* rinvbuf = maxbuf + 16;             // [16]

  const int tid  = threadIdx.x;
  const int wave = tid >> 5;
  const int lane = tid & 31;
  const int row  = lane & 15;
  const int half = lane >> 4;
  const int bid = blockIdx.x;
  const int qt = bid & 127;                 // N/16 tiles
  const int h  = (bid >> 7) % H_;
  const int bb = bid / (128 * H_);
  const int q0 = qt * 16;

  const f16* qhead = qh + (size_t)(bb * H_ + h) * N_ * HD_;
  const f16* khead = kh + (size_t)(bb * H_ + h) * N_ * HD_;
  const f16* vhead = vt + (size_t)(bb * H_ + h) * HD_ * N_;
  const float* rpbh = rpb + ((size_t)h * N_ + q0) * N_;

  // Q A-fragments (HD=64 -> two K=32 fragments); q already scaled by 1/sqrt(d)
  const f16* qrow = qhead + (size_t)(q0 + row) * HD_;
  v16h a0, a1;
  {
    v8h l0 = *(const v8h*)(qrow +      half * 16);
    v8h h0 = *(const v8h*)(qrow + 8 + half * 16);
    v8h l1 = *(const v8h*)(qrow + 32 + half * 16);
    v8h h1 = *(const v8h*)(qrow + 40 + half * 16);
#pragma unroll
    for (int e = 0; e < 8; ++e) {
      a0[e] = l0[e]; a0[e + 8] = h0[e];
      a1[e] = l1[e]; a1[e + 8] = h1[e];
    }
  }

  // ---- Phase 1: S = q.k^T + rpb, into LDS (waves split key tiles) --------
  for (int kt = wave; kt < N_ / 16; kt += 4) {
    const int cb = kt * 16;
    const f16* krow = khead + (size_t)(cb + row) * HD_;  // B: key=col, d=k-dim
    if (kt + 4 < N_ / 16)  // prefetch next K tile -> global_prefetch_b8
      __builtin_prefetch(krow + 64 * HD_ + half * 16, 0, 0);
    v16h b0 = *(const v16h*)(krow +      half * 16);
    v16h b1 = *(const v16h*)(krow + 32 + half * 16);
    v8f c = {};
    c = wmma16x16x32(a0, b0, c);
    c = wmma16x16x32(a1, b1, c);
#pragma unroll
    for (int r = 0; r < 8; ++r) {
      const int mr = half * 8 + r;
      S[mr * SROW + cb + row] = c[r] + rpbh[(size_t)mr * N_ + cb + row];
    }
  }
  __syncthreads();

  // ---- Phase 2: row max / sum; each row scanned by 8 lanes (seg) ---------
  const int seg = tid >> 4;                 // 0..7
  const int r2  = tid & 15;                 // row this lane reduces
  const float* Srow = S + r2 * SROW + seg * (N_ / 8);
  float m = -3.0e38f;
  for (int j = 0; j < N_ / 32; ++j) {
    v4f v = *(const v4f*)(Srow + 4 * j);
    m = fmaxf(m, fmaxf(fmaxf(v[0], v[1]), fmaxf(v[2], v[3])));
  }
  pmax[seg * 16 + r2] = m;
  __syncthreads();
#pragma unroll
  for (int s = 0; s < 8; ++s) m = fmaxf(m, pmax[s * 16 + r2]);
  float sum = 0.f;
  for (int j = 0; j < N_ / 32; ++j) {
    v4f v = *(const v4f*)(Srow + 4 * j);
    sum += __expf(v[0] - m) + __expf(v[1] - m) + __expf(v[2] - m) +
           __expf(v[3] - m);
  }
  psum[seg * 16 + r2] = sum;
  __syncthreads();
  float tot = 0.f;
#pragma unroll
  for (int s = 0; s < 8; ++s) tot += psum[s * 16 + r2];
  if (seg == 0) { maxbuf[r2] = m; rinvbuf[r2] = 1.0f / tot; }
  __syncthreads();

  // ---- Phase 3: stream normalized attn to global (non-temporal vec4) ----
  float* aout = attn_out + ((size_t)(bb * H_ + h) * N_ + q0) * N_;
  for (int vi = tid; vi < 16 * N_ / 4; vi += 128) {
    const int rr = vi >> 9;            // (vi*4)/2048
    const int cc = (vi & 511) * 4;
    v4f v = *(const v4f*)(S + rr * SROW + cc);
    const float rm = maxbuf[rr], ri = rinvbuf[rr];
    v4f p;
    p[0] = __expf(v[0] - rm) * ri;
    p[1] = __expf(v[1] - rm) * ri;
    p[2] = __expf(v[2] - rm) * ri;
    p[3] = __expf(v[3] - rm) * ri;
    __builtin_nontemporal_store(p, (v4f*)(aout + (size_t)rr * N_ + cc));
  }

  // ---- Phase 4: out_head = P.V ; wave w owns d-columns [16w, 16w+16) -----
  const float rm = maxbuf[row];
  const float ri = rinvbuf[row];
  v8f acc = {};
  const f16* vb0 = vhead + (size_t)(wave * 16 + row) * N_;
  for (int kt = 0; kt < N_ / 32; ++kt) {
    const int kb = kt * 32;
    // A-fragment of probabilities, exp-normalized on the fly, f32 -> f16
    const float* Sr = S + row * SROW + kb + half * 16;
    v16h p;
#pragma unroll
    for (int e = 0; e < 16; ++e) {
      const int k = e;
      p[e] = (f16)(__expf(Sr[k] - rm) * ri);
    }
    // B-fragment from transposed V: contiguous 16 f16 per lane
    v16h b = *(const v16h*)(vb0 + kb + half * 16);
    acc = wmma16x16x32(p, b, acc);
  }

  // oh is f16 [B,N,C]; this wave writes cols h*64 + wave*16 + row
  f16* orow = oh + ((size_t)bb * N_ + q0) * C_ + h * HD_ + wave * 16 + row;
#pragma unroll
  for (int r = 0; r < 8; ++r)
    orow[(size_t)(half * 8 + r) * C_] = (f16)acc[r];
}

// ---------------------------------------------------------------------------
// Kernel 3: output projection (all-f16 operands, f32 accumulate).
// out[m][n] = oh[m][:] . Wproj[n][:] + bproj[n]
// ---------------------------------------------------------------------------
__global__ __launch_bounds__(128) void proj_kernel(
    const f16* __restrict__ oh, const f16* __restrict__ wprojh,
    const float* __restrict__ bproj, float* __restrict__ out) {
  const int lane = threadIdx.x & 31;
  const int wave = threadIdx.x >> 5;
  const int row  = lane & 15;
  const int half = lane >> 4;
  const int mt = blockIdx.x;                 // 256
  const int nt = blockIdx.y * 4 + wave;      // 24 col tiles
  const int mrow = mt * 16 + row;
  const int col  = nt * 16 + row;
  const f16* orow = oh     + (size_t)mrow * C_;
  const f16* wrow = wprojh + (size_t)col  * C_;

  v8f c = {};
  for (int kb = 0; kb < C_; kb += 32) {
    v8h alo = *(const v8h*)(orow + kb + half * 16);
    v8h ahi = *(const v8h*)(orow + kb + 8 + half * 16);
    v16h a;
#pragma unroll
    for (int e = 0; e < 8; ++e) { a[e] = alo[e]; a[e + 8] = ahi[e]; }
    v16h b = *(const v16h*)(wrow + kb + half * 16);
    c = wmma16x16x32(a, b, c);
  }
  const float bias = bproj[col];
#pragma unroll
  for (int r = 0; r < 8; ++r) {
    const int mrr = mt * 16 + half * 8 + r;
    out[(size_t)mrr * C_ + col] = c[r] + bias;
  }
}

// ---------------------------------------------------------------------------
extern "C" void kernel_launch(void* const* d_in, const int* in_sizes, int n_in,
                              void* d_out, int out_size, void* d_ws,
                              size_t ws_size, hipStream_t stream) {
  const float* x     = (const float*)d_in[0];  // [B,N,C]
  const float* rpb   = (const float*)d_in[1];  // [H,N,N]
  const float* Wqkv  = (const float*)d_in[2];  // [3C,C]
  const float* bqkv  = (const float*)d_in[3];  // [3C]
  const float* Wproj = (const float*)d_in[4];  // [C,C]
  const float* bproj = (const float*)d_in[5];  // [C]

  float* out  = (float*)d_out;                 // [B,N,C]
  float* attn = out + (size_t)B_ * N_ * C_;    // [B,H,N,N]

  // workspace (all f16): xh[B,N,C], wqkvh[3C,C], wprojh[C,C],
  //                      q/k[B,H,N,HD], v^T[B,H,HD,N], oh[B,N,C]
  const size_t XSZ = (size_t)B_ * N_ * C_;     // 1572864
  const size_t WQ  = (size_t)3 * C_ * C_;      // 442368
  const size_t WP  = (size_t)C_ * C_;          // 147456
  const size_t HSZ = (size_t)B_ * H_ * N_ * HD_;
  f16* xh     = (f16*)d_ws;
  f16* wqkvh  = xh + XSZ;
  f16* wprojh = wqkvh + WQ;
  f16* qh     = wprojh + WP;
  f16* kh     = qh + HSZ;
  f16* vt     = kh + HSZ;
  f16* oh     = vt + HSZ;

  cvt_kernel<<<(int)(XSZ / 4 + 255) / 256, 256, 0, stream>>>(x, xh,
                                                             (int)(XSZ / 4));
  cvt_kernel<<<(int)(WQ / 4 + 255) / 256, 256, 0, stream>>>(Wqkv, wqkvh,
                                                            (int)(WQ / 4));
  cvt_kernel<<<(int)(WP / 4 + 255) / 256, 256, 0, stream>>>(Wproj, wprojh,
                                                            (int)(WP / 4));

  dim3 g1((B_ * N_) / 16, (3 * C_) / (16 * 4));
  qkv_kernel<<<g1, 128, 0, stream>>>(xh, wqkvh, bqkv, qh, kh, vt);

  const size_t lds_bytes = (size_t)(16 * SROW + 8 * 16 * 2 + 32) * sizeof(float);
  attn_kernel<<<B_ * H_ * (N_ / 16), 128, lds_bytes, stream>>>(qh, kh, vt, rpb,
                                                               attn, oh);

  dim3 g3((B_ * N_) / 16, C_ / (16 * 4));
  proj_kernel<<<g3, 128, 0, stream>>>(oh, wprojh, bproj, out);
}
